// BatteryMoEFlattenIntraCycleMoELayer_29686813950420
// MI455X (gfx1250) — matrix operand order repaired and checked by reading, not compiled, not run on hardware
//
#include <hip/hip_runtime.h>
#include <hip/hip_bf16.h>
#include <stdint.h>

// ---------------------------------------------------------------------------
// BatteryMoE top-2 MoE layer for MI455X (gfx1250), wave32 + WMMA bf16.
//
// Sizes (fixed by the reference): B=16384, E=8, D_IN=900, D_MODEL=512, D_SEL=128
// Strategy: expert-bucketed grouped GEMM over exactly 2*B assignments,
// bf16 WMMA (v_wmma_f32_16x16x32_bf16), outputs combined with
// global_atomic_pk_add_bf16 (each row has exactly 2 expert contributions).
// This round: B staging in 4x2 micro-blocks -> ds_store_b64 + fully-packed
// v_cvt_pk_bf16_f32 (3x fewer DS stores, half the conversion VALU).
// ---------------------------------------------------------------------------

#define NB      16384
#define NE      8
#define DIN     900
#define DMODEL  512
#define DSEL    128
#define MT      32            // rows per GEMM tile (gathered, single expert)
#define NT      128           // cols per GEMM tile (8 waves x 16)
#define KT      32            // bf16 WMMA K
#define NFULL   (DIN / KT)    // 28 guard-free K tiles; 1 guarded tail tile
#define CAP     16416         // per-expert bucket capacity (>= B, +pad room)
#define MAXTILES 1032         // 2*B/MT + NE  (static upper bound)
#define LSTR    40            // LDS row stride in bf16 elems (80B: 16B-aligned)
#define EPSV    1e-9f

typedef __attribute__((ext_vector_type(16))) __bf16 v16bf;
typedef __attribute__((ext_vector_type(8)))  float  v8f;

union Frag16 { uint4 q[2]; v16bf v; };

__device__ __forceinline__ unsigned pack_bf16x2(float a, float b) {
    __bf16 x = (__bf16)a, y = (__bf16)b;
    unsigned short ux = __builtin_bit_cast(unsigned short, x);
    unsigned short uy = __builtin_bit_cast(unsigned short, y);
    return (unsigned)ux | ((unsigned)uy << 16);
}

// ---------------------------------------------------------------------------
// Kernel 1: zero small workspace state
// ---------------------------------------------------------------------------
__global__ void k_init(float* s_accum, unsigned* counts, unsigned* ntiles) {
    int t = threadIdx.x;
    if (t == 0) { *s_accum = 0.0f; *ntiles = 0u; }
    if (t < NE) counts[t] = 0u;
}

// ---------------------------------------------------------------------------
// Kernel 2: zero final_out region of d_out (bf16 accumulated via atomics)
// ---------------------------------------------------------------------------
__global__ __launch_bounds__(256) void k_zero_out(uint4* p) {
    unsigned i = blockIdx.x * blockDim.x + threadIdx.x;
    if (i < (NB * DMODEL * 2u) / 16u) p[i] = uint4{0u, 0u, 0u, 0u};
}

// ---------------------------------------------------------------------------
// Kernel 3: per-row softmax / selection embedding / top-2 gates -> buckets
// ---------------------------------------------------------------------------
__global__ __launch_bounds__(128) void k_phase1(
    const float* __restrict__ logits, const int* __restrict__ masks,
    const float* __restrict__ selemb, float* __restrict__ sel_out,
    float* s_accum, unsigned* counts, int* brows, float* bgates)
{
    int b = blockIdx.x;
    int t = threadIdx.x;
    __shared__ float sl[NE];
    __shared__ int   sm[NE];
    if (t < NE) { sl[t] = logits[b * NE + t]; sm[t] = masks[b * NE + t]; }
    __syncthreads();

    float raw[NE];
    float mx = sl[0];
    #pragma unroll
    for (int e = 1; e < NE; ++e) mx = fmaxf(mx, sl[e]);
    float sum = 0.0f;
    #pragma unroll
    for (int e = 0; e < NE; ++e) { raw[e] = expf(sl[e] - mx); sum += raw[e]; }
    float inv = 1.0f / sum;
    #pragma unroll
    for (int e = 0; e < NE; ++e) raw[e] *= inv;

    float wi[NE];
    float si = 0.0f;
    #pragma unroll
    for (int e = 0; e < NE; ++e) { wi[e] = (sm[e] == 1) ? 0.0f : raw[e]; si += wi[e]; }
    float winv = 1.0f / (si + EPSV);
    float acc = 0.0f;
    #pragma unroll
    for (int e = 0; e < NE; ++e)
        acc += wi[e] * winv * selemb[((size_t)b * NE + e) * DSEL + t];
    sel_out[(size_t)b * DSEL + t] = acc;

    if (t == 0) {
        float act[NE];
        float s = 0.0f;
        #pragma unroll
        for (int e = 0; e < NE; ++e) { act[e] = (sm[e] == 1) ? raw[e] : 0.0f; s += act[e]; }
        atomicAdd(s_accum, s);

        int i1 = 0;
        #pragma unroll
        for (int e = 1; e < NE; ++e) if (act[e] > act[i1]) i1 = e;
        int i2 = (i1 == 0) ? 1 : 0;
        #pragma unroll
        for (int e = 0; e < NE; ++e) if (e != i1 && act[e] > act[i2]) i2 = e;
        float denom = act[i1] + act[i2] + EPSV;
        float g1 = act[i1] / denom, g2 = act[i2] / denom;

        unsigned p1 = atomicAdd(&counts[i1], 1u);
        brows[i1 * CAP + p1] = b;  bgates[i1 * CAP + p1] = g1;
        unsigned p2 = atomicAdd(&counts[i2], 1u);
        brows[i2 * CAP + p2] = b;  bgates[i2 * CAP + p2] = g2;
    }
}

// ---------------------------------------------------------------------------
// Kernel 4: pad buckets to multiples of MT, build tile descriptors
// ---------------------------------------------------------------------------
__global__ __launch_bounds__(256) void k_pad_desc(
    unsigned* counts, unsigned* ntiles, int2* desc, int* brows, float* bgates)
{
    __shared__ unsigned tl[NE], base[NE + 1];
    int t = threadIdx.x;
    if (t < NE) {
        unsigned c  = counts[t];
        unsigned pc = (c + (MT - 1u)) & ~(MT - 1u);
        for (unsigned i = c; i < pc; ++i) { brows[t * CAP + i] = 0; bgates[t * CAP + i] = 0.0f; }
        tl[t] = pc / MT;
    }
    __syncthreads();
    if (t == 0) {
        unsigned s = 0;
        for (int e = 0; e < NE; ++e) { base[e] = s; s += tl[e]; }
        base[NE] = s;
        *ntiles = s;
    }
    __syncthreads();
    unsigned total = base[NE];
    for (unsigned idx = t; idx < total; idx += blockDim.x) {
        int e = 0;
        while (!(idx >= base[e] && idx < base[e + 1])) ++e;
        unsigned loc = idx - base[e];
        desc[idx] = make_int2(e, e * CAP + (int)(loc * MT));
    }
}

// ---------------------------------------------------------------------------
// Kernel 5: finalize guide loss
// ---------------------------------------------------------------------------
__global__ void k_finalize(const float* s_accum, float* gl) {
    float s = *s_accum / (float)NB;
    float d = 1.0f - s;
    *gl = d * d;
}

// ---------------------------------------------------------------------------
// Kernel 6: grouped GEMM, software-pipelined.
//   Block = 32 gathered rows (one expert) x 128 cols; 8 waves; wave w owns
//   cols [w*16,w*16+16) with two 16x16 f32 accumulators (row halves).
//   28 guard-free K tiles + 1 guarded tail tile. B staged in 4x2 blocks.
// ---------------------------------------------------------------------------
__global__ __launch_bounds__(256) void k_gemm(
    const float* __restrict__ x, const float* __restrict__ W,
    const float* __restrict__ bias, __hip_bfloat16* __restrict__ out,
    const unsigned* __restrict__ ntiles, const int2* __restrict__ desc,
    const int* __restrict__ brows, const float* __restrict__ bgates)
{
    if (blockIdx.x >= *ntiles) return;
    int2 dsc = desc[blockIdx.x];
    const int e   = dsc.x;
    const int seg = dsc.y;
    const int n0  = blockIdx.y * NT;

    __shared__ __bf16 Alds[MT * LSTR];   // [row][k], stride 40 bf16 (80 B)
    __shared__ __bf16 Blds[NT * LSTR];   // [col][k] (transposed), stride 40
    __shared__ int    sRow[MT];
    __shared__ float  sGate[MT];

    const int tid  = threadIdx.x;
    const int wave = tid >> 5;
    const int lane = tid & 31;

    if (tid < MT) { sRow[tid] = brows[seg + tid]; sGate[tid] = bgates[seg + tid]; }
    __syncthreads();

    const float* We = W + (size_t)e * DIN * DMODEL;

    // ---- loop-invariant staging geometry -----------------------------------
    // A: thread covers pairs (row ar0, k akk..akk+1) and (row ar0+16, same k)
    const int ar0 = tid >> 4;            // 0..15
    const int akk = (tid & 15) * 2;      // 0..30 (even)
    const int ar1 = ar0 + 16;
    const float* xa0 = x + (size_t)sRow[ar0] * DIN + akk;
    const float* xa1 = x + (size_t)sRow[ar1] * DIN + akk;
    unsigned* aSt0 = (unsigned*)&Alds[ar0 * LSTR + akk];
    unsigned* aSt1 = (unsigned*)&Alds[ar1 * LSTR + akk];
    // B: 2 items j, each a 4x2 block: k = 4*bkq + 16*j + (0..3), cols bnn..bnn+1
    const int bkq = tid >> 6;            // 0..3  (k base = 4*bkq)
    const int bnn = (tid & 63) * 2;      // 0..126 (even)
    const float* wb = We + (size_t)(4 * bkq) * DMODEL + n0 + bnn;
    uint2* bSt0 = (uint2*)&Blds[bnn * LSTR + 4 * bkq];        // 8B aligned
    uint2* bSt1 = (uint2*)&Blds[(bnn + 1) * LSTR + 4 * bkq];  // 8B aligned

    // ---- loop-invariant fragment pointers (ISA A 16x32 / B 32x16 layouts) --
    const int mrow = lane & 15;
    const int ka   = (lane < 16) ? 0 : 8;
    const uint4* fA0a = (const uint4*)&Alds[mrow * LSTR + ka];
    const uint4* fA0b = (const uint4*)&Alds[mrow * LSTR + ka + 16];
    const uint4* fA1a = (const uint4*)&Alds[(mrow + 16) * LSTR + ka];
    const uint4* fA1b = (const uint4*)&Alds[(mrow + 16) * LSTR + ka + 16];
    const int col = wave * 16 + (lane & 15);
    const int kb  = (lane < 16) ? 0 : 8;
    const uint4* fBa = (const uint4*)&Blds[col * LSTR + kb];
    const uint4* fBb = (const uint4*)&Blds[col * LSTR + kb + 16];

    v8f acc0 = {0.f,0.f,0.f,0.f,0.f,0.f,0.f,0.f};
    v8f acc1 = {0.f,0.f,0.f,0.f,0.f,0.f,0.f,0.f};

    // ---- prologue: load k-tile 0 into registers (guard-free) ---------------
    float2 ra0 = *(const float2*)xa0;
    float2 ra1 = *(const float2*)xa1;
    float2 rb[2][4];
    #pragma unroll
    for (int j = 0; j < 2; ++j)
        #pragma unroll
        for (int r = 0; r < 4; ++r)
            rb[j][r] = *(const float2*)(wb + (size_t)(16 * j + r) * DMODEL);

    for (int it = 0; it < NFULL; ++it) {
        // store current registers to LDS (bf16)
        *aSt0 = pack_bf16x2(ra0.x, ra0.y);
        *aSt1 = pack_bf16x2(ra1.x, ra1.y);
        #pragma unroll
        for (int j = 0; j < 2; ++j) {
            bSt0[4 * j] = make_uint2(pack_bf16x2(rb[j][0].x, rb[j][1].x),
                                     pack_bf16x2(rb[j][2].x, rb[j][3].x));
            bSt1[4 * j] = make_uint2(pack_bf16x2(rb[j][0].y, rb[j][1].y),
                                     pack_bf16x2(rb[j][2].y, rb[j][3].y));
        }
        __syncthreads();

        // issue next tile's global loads (overlap with WMMA below)
        const int kn = (it + 1) * KT;
        if (it + 1 < NFULL) {                       // guard-free path
            ra0 = *(const float2*)(xa0 + kn);
            ra1 = *(const float2*)(xa1 + kn);
            #pragma unroll
            for (int j = 0; j < 2; ++j)
                #pragma unroll
                for (int r = 0; r < 4; ++r)
                    rb[j][r] = *(const float2*)(wb + (size_t)(kn + 16 * j + r) * DMODEL);
        } else {                                    // guarded tail (k0=896)
            ra0 = (kn + akk + 1 < DIN) ? *(const float2*)(xa0 + kn) : float2{0.f, 0.f};
            ra1 = (kn + akk + 1 < DIN) ? *(const float2*)(xa1 + kn) : float2{0.f, 0.f};
            #pragma unroll
            for (int j = 0; j < 2; ++j) {
                const bool ok = (kn + 4 * bkq + 16 * j + 3 < DIN);  // 4-row block all-or-nothing
                #pragma unroll
                for (int r = 0; r < 4; ++r)
                    rb[j][r] = ok ? *(const float2*)(wb + (size_t)(kn + 16 * j + r) * DMODEL)
                                  : float2{0.f, 0.f};
            }
        }

        // fragments + WMMA for current tile
        Frag16 fa0, fa1, fb;
        fa0.q[0] = *fA0a; fa0.q[1] = *fA0b;
        fa1.q[0] = *fA1a; fa1.q[1] = *fA1b;
        fb.q[0]  = *fBa;  fb.q[1]  = *fBb;
        acc0 = __builtin_amdgcn_wmma_f32_16x16x32_bf16(
                   false, fa0.v, false, fb.v, (short)0, acc0, false, false);
        acc1 = __builtin_amdgcn_wmma_f32_16x16x32_bf16(
                   false, fa1.v, false, fb.v, (short)0, acc1, false, false);
        __syncthreads();
    }

    // ---- epilogue tile (k0 = 896, zero-padded registers) -------------------
    *aSt0 = pack_bf16x2(ra0.x, ra0.y);
    *aSt1 = pack_bf16x2(ra1.x, ra1.y);
    #pragma unroll
    for (int j = 0; j < 2; ++j) {
        bSt0[4 * j] = make_uint2(pack_bf16x2(rb[j][0].x, rb[j][1].x),
                                 pack_bf16x2(rb[j][2].x, rb[j][3].x));
        bSt1[4 * j] = make_uint2(pack_bf16x2(rb[j][0].y, rb[j][1].y),
                                 pack_bf16x2(rb[j][2].y, rb[j][3].y));
    }
    __syncthreads();
    {
        Frag16 fa0, fa1, fb;
        fa0.q[0] = *fA0a; fa0.q[1] = *fA0b;
        fa1.q[0] = *fA1a; fa1.q[1] = *fA1b;
        fb.q[0]  = *fBa;  fb.q[1]  = *fBb;
        acc0 = __builtin_amdgcn_wmma_f32_16x16x32_bf16(
                   false, fa0.v, false, fb.v, (short)0, acc0, false, false);
        acc1 = __builtin_amdgcn_wmma_f32_16x16x32_bf16(
                   false, fa1.v, false, fb.v, (short)0, acc1, false, false);
    }

    // ---- epilogue: out[row, n] += gate*(acc + bias[e,n]) -------------------
    const int n    = n0 + wave * 16 + (lane & 15);
    const float bv = bias[e * DMODEL + n];
    const int mb   = (lane < 16) ? 0 : 8;        // C/D layout: VGPR j -> M=j or j+8
    const int ncol = n & ~1;
    const bool hi  = (n & 1) != 0;

    #pragma unroll
    for (int j = 0; j < 8; ++j) {
        {
            int m = mb + j;
            float g = sGate[m];
            float v = g * (acc0[j] + bv);
            __bf16 h = (__bf16)v;
            unsigned pk = (unsigned)__builtin_bit_cast(unsigned short, h);
            if (hi) pk <<= 16;
            __hip_bfloat16* dst = out + (size_t)sRow[m] * DMODEL + ncol;
            asm volatile("global_atomic_pk_add_bf16 %0, %1, off"
                         :: "v"((unsigned long long)(uintptr_t)dst), "v"(pk)
                         : "memory");
        }
        {
            int m = mb + j + 16;
            float g = sGate[m];
            float v = g * (acc1[j] + bv);
            __bf16 h = (__bf16)v;
            unsigned pk = (unsigned)__builtin_bit_cast(unsigned short, h);
            if (hi) pk <<= 16;
            __hip_bfloat16* dst = out + (size_t)sRow[m] * DMODEL + ncol;
            asm volatile("global_atomic_pk_add_bf16 %0, %1, off"
                         :: "v"((unsigned long long)(uintptr_t)dst), "v"(pk)
                         : "memory");
        }
    }
}

// ---------------------------------------------------------------------------
// Host launcher
// ---------------------------------------------------------------------------
extern "C" void kernel_launch(void* const* d_in, const int* in_sizes, int n_in,
                              void* d_out, int out_size, void* d_ws, size_t ws_size,
                              hipStream_t stream) {
    const float* x      = (const float*)d_in[0];   // [B, 900]
    const float* logits = (const float*)d_in[1];   // [B, 8]
    const int*   masks  = (const int*)  d_in[2];   // [B, 8]
    const float* selemb = (const float*)d_in[3];   // [B, 8, 128]
    const float* W      = (const float*)d_in[4];   // [8, 900, 512]
    const float* bias   = (const float*)d_in[5];   // [8, 512]

    // d_out byte layout (tuple outputs concatenated in return order):
    //   [0 .. B*512*2)                : final_out bf16
    //   [B*512*2 .. +4)               : guide_loss f32
    //   [B*512*2+4 .. +B*128*4)       : selection_embedding f32
    char* ob = (char*)d_out;
    __hip_bfloat16* outp = (__hip_bfloat16*)ob;
    float* gl     = (float*)(ob + (size_t)NB * DMODEL * 2);
    float* selout = (float*)(ob + (size_t)NB * DMODEL * 2 + 4);

    // workspace layout (~1.04 MB total)
    char* ws = (char*)d_ws;
    float*    s_accum = (float*)ws;
    unsigned* counts  = (unsigned*)(ws + 64);
    unsigned* ntiles  = (unsigned*)(ws + 96);
    int2*     desc    = (int2*)(ws + 128);
    int*      brows   = (int*)(ws + 16384);
    float*    bgates  = (float*)(ws + 16384 + (size_t)NE * CAP * 4);

    k_init<<<1, 64, 0, stream>>>(s_accum, counts, ntiles);
    k_zero_out<<<(NB * DMODEL * 2 / 16 + 255) / 256, 256, 0, stream>>>((uint4*)d_out);
    k_phase1<<<NB, 128, 0, stream>>>(logits, masks, selemb, selout,
                                     s_accum, counts, brows, bgates);
    k_pad_desc<<<1, 256, 0, stream>>>(counts, ntiles, desc, brows, bgates);
    k_finalize<<<1, 1, 0, stream>>>(s_accum, gl);
    dim3 g(MAXTILES, DMODEL / NT);
    k_gemm<<<g, 256, 0, stream>>>(x, W, bias, outp, ntiles, desc, brows, bgates);
}
